// AdaptiveJacobianPrunedViT_35313221108257
// MI455X (gfx1250) — matrix operand order repaired and checked by reading, not compiled, not run on hardware
//
#include <hip/hip_runtime.h>
#include <math.h>

#define B_      64
#define T_      197
#define SP_     208     /* padded attention-score row stride (multiple of 16) */
#define D_      768
#define H_      12
#define DH_     64
#define MLP_    3072
#define NCLS_   1000
#define NPATCH_ 196
#define GAMMA_  0.05f
#define MIN_TOK_ 16
#define EPS_    1e-6f
#define LN_EPS_ 1e-6f
#define SCALE_  0.125f  /* 64^-0.5 */

typedef __attribute__((ext_vector_type(16))) __bf16 v16bf;
typedef __attribute__((ext_vector_type(8)))  __bf16 v8bf;
typedef __attribute__((ext_vector_type(4)))  __bf16 v4bf;
typedef __attribute__((ext_vector_type(8)))  float  v8f;

__device__ __forceinline__ float gelu_exact(float x) {
  return 0.5f * x * (1.0f + erff(x * 0.70710678118654752440f));
}

__device__ __forceinline__ float block_reduce_sum(float v, float* red) {
  int tid = threadIdx.x;
  red[tid] = v; __syncthreads();
  for (int s = 128; s > 0; s >>= 1) { if (tid < s) red[tid] += red[tid + s]; __syncthreads(); }
  float r = red[0]; __syncthreads();
  return r;
}

// ---------------------------------------------------------------------------
// Generic batched bf16-WMMA GEMM:  C[z] = act( A[z] @ B[z] + bias + Res[z] )
// A: f32, row stride rsA, contiguous in k.  B: f32 at [k*kstrB + n*nstrB]
// (one of kstrB/nstrB must be 1).  Batch base = z/zdiv*bs?o + z%zdiv*bs?i + off?.
// Runtime dims via nullable device pointers dM/dN/dK.
// Tile: 64x128, 8 waves, each wave 16x64 -> 4 chained v_wmma_f32_16x16x32_bf16.
// Staging has a block-uniform (scalar-branch) fast path for interior tiles.
// ---------------------------------------------------------------------------
#define TM 64
#define TN 128
#define TK 32
#define LDK 40   /* padded LDS k-stride: 80B rows keep b64/b128 alignment */

__global__ __launch_bounds__(256) void wmma_gemm_kernel(
    const float* __restrict__ A, const float* __restrict__ Bm,
    const float* __restrict__ bias, const float* __restrict__ Res,
    float* __restrict__ C,
    int M, int N, int K,
    const int* __restrict__ dM, const int* __restrict__ dN, const int* __restrict__ dK,
    int zdiv,
    long long offA, long long bsAo, long long bsAi, int rsA,
    long long offB, long long bsBo, long long bsBi, int kstrB, int nstrB,
    long long offC, long long bsCo, long long bsCi, int rsC,
    long long offR, long long bsRo, long long bsRi, int rsR,
    int act)
{
  __shared__ __bf16 As[TM][LDK];    // [m][k]
  __shared__ __bf16 Bst[TN][LDK];   // transposed: [n][k]

  if (dM) M = *dM;
  if (dN) N = *dN;
  if (dK) K = *dK;

  int m0 = blockIdx.x * TM;
  int n0 = blockIdx.y * TN;
  if (m0 >= M || n0 >= N) return;

  int z  = blockIdx.z;
  int zo = z / zdiv, zi = z % zdiv;

  const float* Ab = A  + offA + (long long)zo * bsAo + (long long)zi * bsAi;
  const float* Bb = Bm + offB + (long long)zo * bsBo + (long long)zi * bsBi;
  float*       Cb = C  + offC + (long long)zo * bsCo + (long long)zi * bsCi;
  const float* Rb = Res ? (Res + offR + (long long)zo * bsRo + (long long)zi * bsRi) : nullptr;

  int tid  = threadIdx.x;
  int lane = tid & 31;
  int wave = tid >> 5;     // 0..7
  int wr   = wave >> 1;    // 0..3  (16-row strip)
  int wc   = wave & 1;     // 0..1  (64-col strip)

  v8f acc0 = {}, acc1 = {}, acc2 = {}, acc3 = {};

  const bool bNmaj = (nstrB == 1);       // B contiguous along n (weights, A@V)
  const bool fullM = (m0 + TM <= M);     // block-uniform
  const bool fullN = (n0 + TN <= N);     // block-uniform

  // thread's fixed chunk coordinates
  const int ar0 = tid >> 3,          ac0 = (tid & 7) * 4;          // A chunk 0
  const int ar1 = (tid + 256) >> 3,  ac1 = ((tid + 256) & 7) * 4;  // A chunk 1

  for (int k0 = 0; k0 < K; k0 += TK) {
    const bool fullK = (k0 + TK <= K);   // block-uniform

    // ---- preload A chunks (2 per thread) into registers ----
    float4 fa[2];
    if (fullM && fullK) {                 // scalar-branch fast path
      fa[0] = *(const float4*)(Ab + (long long)(m0 + ar0) * rsA + (k0 + ac0));
      fa[1] = *(const float4*)(Ab + (long long)(m0 + ar1) * rsA + (k0 + ac1));
    } else {
#pragma unroll
      for (int j = 0; j < 2; ++j) {
        int r = j ? ar1 : ar0, c4 = j ? ac1 : ac0;
        int gm = m0 + r, gk = k0 + c4;
        float4 v = make_float4(0.f, 0.f, 0.f, 0.f);
        if (gm < M) {
          if (gk + 3 < K) {
            v = *(const float4*)(Ab + (long long)gm * rsA + gk);
          } else {
            if (gk + 0 < K) v.x = Ab[(long long)gm * rsA + gk + 0];
            if (gk + 1 < K) v.y = Ab[(long long)gm * rsA + gk + 1];
            if (gk + 2 < K) v.z = Ab[(long long)gm * rsA + gk + 2];
          }
        }
        fa[j] = v;
      }
    }

    // ---- preload B chunks (4 per thread) ----
    float4 fb[4];
    if (bNmaj) {
      if (fullN && fullK) {
#pragma unroll
        for (int j = 0; j < 4; ++j) {
          int id = tid + j * 256;
          int r = id >> 5, c4 = (id & 31) * 4;
          fb[j] = *(const float4*)(Bb + (long long)(k0 + r) * kstrB + (n0 + c4));
        }
      } else {
#pragma unroll
        for (int j = 0; j < 4; ++j) {
          int id = tid + j * 256;
          int r = id >> 5, c4 = (id & 31) * 4;
          int gk = k0 + r, gn = n0 + c4;
          float4 v = make_float4(0.f, 0.f, 0.f, 0.f);
          if (gk < K) {
            if (gn + 3 < N) {
              v = *(const float4*)(Bb + (long long)gk * kstrB + gn);
            } else {
              if (gn + 0 < N) v.x = Bb[(long long)gk * kstrB + gn + 0];
              if (gn + 1 < N) v.y = Bb[(long long)gk * kstrB + gn + 1];
              if (gn + 2 < N) v.z = Bb[(long long)gk * kstrB + gn + 2];
            }
          }
          fb[j] = v;
        }
      }
    } else {
      if (fullN && fullK) {
#pragma unroll
        for (int j = 0; j < 4; ++j) {
          int id = tid + j * 256;
          int r = id >> 3, c4 = (id & 7) * 4;
          fb[j] = *(const float4*)(Bb + (long long)(n0 + r) * nstrB + (k0 + c4));
        }
      } else {
#pragma unroll
        for (int j = 0; j < 4; ++j) {
          int id = tid + j * 256;
          int r = id >> 3, c4 = (id & 7) * 4;
          int gn = n0 + r, gk = k0 + c4;
          float4 v = make_float4(0.f, 0.f, 0.f, 0.f);
          if (gn < N) {
            if (gk + 3 < K) {
              v = *(const float4*)(Bb + (long long)gn * nstrB + gk);
            } else {
              if (gk + 0 < K) v.x = Bb[(long long)gn * nstrB + gk + 0];
              if (gk + 1 < K) v.y = Bb[(long long)gn * nstrB + gk + 1];
              if (gk + 2 < K) v.z = Bb[(long long)gn * nstrB + gk + 2];
            }
          }
          fb[j] = v;
        }
      }
    }

    // prefetch next k-step (gfx1250 global_prefetch path)
    if (k0 + TK < K) {
      int gm = m0 + (tid & 63);
      if (gm < M) __builtin_prefetch(&Ab[(long long)gm * rsA + (k0 + TK)], 0, 1);
    }
    __syncthreads();   // previous iteration's LDS consumers are done

    // ---- store staged tiles as bf16 ----
    {
      v4bf p0 = { (__bf16)fa[0].x, (__bf16)fa[0].y, (__bf16)fa[0].z, (__bf16)fa[0].w };
      *(v4bf*)&As[ar0][ac0] = p0;
      v4bf p1 = { (__bf16)fa[1].x, (__bf16)fa[1].y, (__bf16)fa[1].z, (__bf16)fa[1].w };
      *(v4bf*)&As[ar1][ac1] = p1;
    }
#pragma unroll
    for (int j = 0; j < 4; ++j) {
      int id = tid + j * 256;
      if (bNmaj) {
        int r = id >> 5, c4 = (id & 31) * 4;
        Bst[c4 + 0][r] = (__bf16)fb[j].x;
        Bst[c4 + 1][r] = (__bf16)fb[j].y;
        Bst[c4 + 2][r] = (__bf16)fb[j].z;
        Bst[c4 + 3][r] = (__bf16)fb[j].w;
      } else {
        int r = id >> 3, c4 = (id & 7) * 4;
        v4bf p = { (__bf16)fb[j].x, (__bf16)fb[j].y, (__bf16)fb[j].z, (__bf16)fb[j].w };
        *(v4bf*)&Bst[r][c4] = p;
      }
    }
    __syncthreads();

    // ---- fragments via wide LDS loads ----
    // A/B use contiguous k order here (k = 16h + s): lane half selects kb 0/16; VGPR4-7 hold kb+8..kb+15.
    int am = wr * 16 + (lane & 15);
    int kb = (lane < 16) ? 0 : 16;
    v8bf a0 = *(const v8bf*)&As[am][kb];
    v8bf a1 = *(const v8bf*)&As[am][kb + 8];
    v16bf af = __builtin_shufflevector(a0, a1, 0,1,2,3,4,5,6,7,8,9,10,11,12,13,14,15);

    int kb2 = (lane < 16) ? 0 : 16;
    int bnb = wc * 64 + (lane & 15);
    {
      v8bf b0 = *(const v8bf*)&Bst[bnb +  0][kb2];
      v8bf b1 = *(const v8bf*)&Bst[bnb +  0][kb2 + 8];
      v16bf bv = __builtin_shufflevector(b0, b1, 0,1,2,3,4,5,6,7,8,9,10,11,12,13,14,15);
      acc0 = __builtin_amdgcn_wmma_f32_16x16x32_bf16(false, af, false, bv, (short)0, acc0, false, false);
    }
    {
      v8bf b0 = *(const v8bf*)&Bst[bnb + 16][kb2];
      v8bf b1 = *(const v8bf*)&Bst[bnb + 16][kb2 + 8];
      v16bf bv = __builtin_shufflevector(b0, b1, 0,1,2,3,4,5,6,7,8,9,10,11,12,13,14,15);
      acc1 = __builtin_amdgcn_wmma_f32_16x16x32_bf16(false, af, false, bv, (short)0, acc1, false, false);
    }
    {
      v8bf b0 = *(const v8bf*)&Bst[bnb + 32][kb2];
      v8bf b1 = *(const v8bf*)&Bst[bnb + 32][kb2 + 8];
      v16bf bv = __builtin_shufflevector(b0, b1, 0,1,2,3,4,5,6,7,8,9,10,11,12,13,14,15);
      acc2 = __builtin_amdgcn_wmma_f32_16x16x32_bf16(false, af, false, bv, (short)0, acc2, false, false);
    }
    {
      v8bf b0 = *(const v8bf*)&Bst[bnb + 48][kb2];
      v8bf b1 = *(const v8bf*)&Bst[bnb + 48][kb2 + 8];
      v16bf bv = __builtin_shufflevector(b0, b1, 0,1,2,3,4,5,6,7,8,9,10,11,12,13,14,15);
      acc3 = __builtin_amdgcn_wmma_f32_16x16x32_bf16(false, af, false, bv, (short)0, acc3, false, false);
    }
    __syncthreads();
  }

  // ---- epilogue: C 16x16 f32 layout (VGPR r -> M=r / r+8 by lane half) ----
  int mbase = m0 + wr * 16 + ((lane >> 4) << 3);
#pragma unroll
  for (int q = 0; q < 4; ++q) {
    v8f accq = (q == 0) ? acc0 : (q == 1) ? acc1 : (q == 2) ? acc2 : acc3;
    int nc = n0 + wc * 64 + q * 16 + (lane & 15);
    if (nc >= N) continue;
    float bv = bias ? bias[nc] : 0.0f;
#pragma unroll
    for (int r = 0; r < 8; ++r) {
      int gm = mbase + r;
      if (gm >= M) continue;
      float v = accq[r] + bv;
      if (Rb) v += Rb[(long long)gm * rsR + nc];
      if (act == 1) v = gelu_exact(v);
      Cb[(long long)gm * rsC + nc] = v;
    }
  }
}

// ---------------------------------------------------------------------------
// Elementwise / reduction kernels (f32, memory-bound)
// ---------------------------------------------------------------------------
__global__ __launch_bounds__(256) void patchify_kernel(const float* __restrict__ x,
                                                       float* __restrict__ pe)
{
  int p = blockIdx.x, b = blockIdx.y;
  int gy = p / 14, gx = p % 14;
  for (int k = threadIdx.x; k < D_; k += 256) {
    int c = k >> 8, rem = k & 255, py = rem >> 4, px = rem & 15;
    pe[((long long)b * NPATCH_ + p) * D_ + k] =
        x[(((long long)b * 3 + c) * 224 + (gy * 16 + py)) * 224 + (gx * 16 + px)];
  }
}

__global__ __launch_bounds__(256) void clspos_kernel(float* __restrict__ tok,
                                                     const float* __restrict__ cls,
                                                     const float* __restrict__ pos)
{
  int t = blockIdx.x, b = blockIdx.y;
  float* row = tok + ((long long)b * T_ + t) * D_;
  for (int d = threadIdx.x; d < D_; d += 256) {
    if (t == 0) row[d] = cls[d] + pos[d];
    else        row[d] += pos[(long long)t * D_ + d];
  }
}

__global__ __launch_bounds__(256) void layernorm_kernel(const float* __restrict__ X,
                                                        float* __restrict__ Y,
                                                        const float* __restrict__ w,
                                                        const float* __restrict__ bta,
                                                        const int* __restrict__ pN)
{
  __shared__ float red[256];
  int ntok = *pN;
  int t = blockIdx.x; if (t >= ntok) return;
  int b = blockIdx.y;
  const float* xr = X + ((long long)b * T_ + t) * D_;
  float*       yr = Y + ((long long)b * T_ + t) * D_;
  int tid = threadIdx.x;
  float s = 0.f;
  for (int d = tid; d < D_; d += 256) s += xr[d];
  s = block_reduce_sum(s, red);
  float mu = s / (float)D_;
  float v = 0.f;
  for (int d = tid; d < D_; d += 256) { float dd = xr[d] - mu; v += dd * dd; }
  v = block_reduce_sum(v, red);
  float rstd = rsqrtf(v / (float)D_ + LN_EPS_);
  for (int d = tid; d < D_; d += 256) yr[d] = (xr[d] - mu) * rstd * w[d] + bta[d];
}

__global__ __launch_bounds__(256) void softmax_kernel(float* __restrict__ S,
                                                      const int* __restrict__ pN)
{
  __shared__ float red[256];
  int ntok = *pN;
  int t = blockIdx.x; if (t >= ntok) return;
  int h = blockIdx.y, b = blockIdx.z;
  float* row = S + (((long long)(b * H_ + h)) * T_ + t) * SP_;
  int tid = threadIdx.x;
  float m = -3.0e38f;
  for (int j = tid; j < ntok; j += 256) m = fmaxf(m, row[j] * SCALE_);
  red[tid] = m; __syncthreads();
  for (int s = 128; s > 0; s >>= 1) { if (tid < s) red[tid] = fmaxf(red[tid], red[tid + s]); __syncthreads(); }
  m = red[0]; __syncthreads();
  float sum = 0.f;
  for (int j = tid; j < ntok; j += 256) { float e = expf(row[j] * SCALE_ - m); row[j] = e; sum += e; }
  sum = block_reduce_sum(sum, red);
  float inv = 1.0f / sum;
  for (int j = tid; j < ntok; j += 256) row[j] *= inv;
}

__global__ __launch_bounds__(256) void aclsmean_kernel(const float* __restrict__ S,
                                                       float* __restrict__ Acls,
                                                       const int* __restrict__ pN)
{
  int b = blockIdx.x;
  int ntok = *pN;
  for (int j = threadIdx.x; j < ntok; j += 256) {
    float s = 0.f;
    for (int h = 0; h < H_; ++h)
      s += S[(((long long)(b * H_ + h)) * T_ + 0) * SP_ + j];
    Acls[(long long)b * T_ + j] = s / (float)H_;
  }
}

__global__ __launch_bounds__(64) void vmean_kernel(const float* __restrict__ qkv,
                                                   float* __restrict__ vmean,
                                                   const int* __restrict__ pN)
{
  int ntok = *pN;
  int t = blockIdx.x; if (t >= ntok) return;
  int b = blockIdx.y;
  int d = threadIdx.x;
  const float* base = qkv + ((long long)b * T_ + t) * (3 * D_) + 2 * D_;
  float s = 0.f;
  for (int h = 0; h < H_; ++h) s += base[h * DH_ + d];
  vmean[((long long)b * T_ + t) * DH_ + d] = s / (float)H_;
}

__global__ __launch_bounds__(256) void batchstats_kernel(const float* __restrict__ vmean,
                                                         const float* __restrict__ Acls,
                                                         float* __restrict__ Jbuf,
                                                         float* __restrict__ sumJ,
                                                         float* __restrict__ clsn,
                                                         const int* __restrict__ pN)
{
  __shared__ float sVm[DH_];
  __shared__ float red[256];
  int b = blockIdx.x;
  int ntok = *pN, N = ntok - 1;
  int tid = threadIdx.x;

  if (tid < DH_) {
    float s = 0.f;
    for (int j = 1; j <= N; ++j) s += vmean[((long long)b * T_ + j) * DH_ + tid];
    sVm[tid] = s / (float)N;
  }
  float c = 0.f;
  if (tid < DH_) { float vv = vmean[((long long)b * T_ + 0) * DH_ + tid]; c = vv * vv; }
  float cn = block_reduce_sum(c, red);
  if (tid == 0) clsn[b] = sqrtf(cn);
  __syncthreads();

  float vn = 0.f;
  if (tid < N) {
    int j = tid + 1;
    float ss = 0.f;
    for (int d = 0; d < DH_; ++d) {
      float dd = vmean[((long long)b * T_ + j) * DH_ + d] - sVm[d];
      ss += dd * dd;
    }
    vn = sqrtf(ss);
  }
  float mu = block_reduce_sum((tid < N) ? vn : 0.f, red) / (float)N;
  float dv = (tid < N) ? (vn - mu) : 0.f;
  float var = block_reduce_sum(dv * dv, red) / (float)(N - 1);
  float sd = sqrtf(var);

  float Jv = 0.f;
  if (tid < N) {
    float g = (vn - mu) / (sd + EPS_);
    if (g < 0.f) g = 0.f;
    Jv = Acls[(long long)b * T_ + (tid + 1)] * g;
    Jbuf[(long long)b * T_ + tid] = Jv;
  }
  float sj = block_reduce_sum(Jv, red);
  if (tid == 0) sumJ[b] = sj;
}

__global__ __launch_bounds__(256) void scalars_kernel(const float* __restrict__ Acls,
                                                      const float* __restrict__ clsn,
                                                      const float* __restrict__ sumJ,
                                                      const float* __restrict__ Jbuf,
                                                      float* __restrict__ scal,
                                                      float* __restrict__ score,
                                                      const int* __restrict__ pN)
{
  __shared__ float red[256];
  int tid = threadIdx.x;
  int N = *pN - 1;
  float r = (tid < B_) ? (1.0f + Acls[(long long)tid * T_] * clsn[tid]) : 0.f;
  float rho = block_reduce_sum(r, red) / (float)B_;
  float mc = (tid < B_) ? sumJ[tid] : 0.f;
  float mass = block_reduce_sum(mc, red) / (float)B_;
  if (tid == 0) { scal[0] = rho; scal[1] = mass; }
  for (int j = tid; j < N; j += 256) {
    float s = 0.f;
    for (int b = 0; b < B_; ++b) s += Jbuf[(long long)b * T_ + j];
    score[j] = s / (float)B_;
  }
}

__global__ __launch_bounds__(256) void select_kernel(const float* __restrict__ scal,
                                                     const float* __restrict__ score,
                                                     int* __restrict__ keep,
                                                     int* __restrict__ pN,
                                                     int* __restrict__ pPrevValid,
                                                     float* __restrict__ pPrevMass)
{
  __shared__ int   kflag[256];
  __shared__ float ssc[256];
  int tid = threadIdx.x;
  int ntok = *pN, N = ntok - 1;

  if (N <= MIN_TOK_) {            // reference: continue (no prev_mass update)
    if (tid < ntok) keep[tid] = tid;
    return;
  }

  float rho = scal[0], mass = scal[1];
  int Nn;
  if (*pPrevValid) {
    float eta = mass / (*pPrevMass + EPS_);
    float cr = rho * eta;
    cr = fminf(fmaxf(cr, 0.25f), 4.0f);
    float kr = powf(cr, -GAMMA_);
    Nn = (int)((float)N * kr);
    if (Nn < MIN_TOK_) Nn = MIN_TOK_;
    if (Nn > N) Nn = N;
  } else {
    Nn = N;
  }

  if (Nn < N) {
    ssc[tid] = (tid < N) ? score[tid] : -3.0e38f;
    __syncthreads();
    int kept = 0;
    if (tid < N) {
      float sj = ssc[tid];
      int rank = 0;
      for (int i = 0; i < N; ++i) {
        float si = ssc[i];
        if (si > sj || (si == sj && i < tid)) rank++;
      }
      kept = (rank < Nn) ? 1 : 0;
    }
    kflag[tid] = kept; __syncthreads();
    if (kept) {
      int pos = 1;
      for (int i = 0; i < tid; ++i) pos += kflag[i];
      keep[pos] = tid + 1;
    }
    if (tid == 0) { keep[0] = 0; *pN = Nn + 1; }
  } else {
    if (tid < ntok) keep[tid] = tid;
  }
  __syncthreads();
  if (tid == 0) { *pPrevMass = mass; *pPrevValid = 1; }
}

__global__ __launch_bounds__(256) void gather_kernel(const float* __restrict__ in,
                                                     float* __restrict__ out,
                                                     const int* __restrict__ keep,
                                                     const int* __restrict__ pN)
{
  int ntok = *pN;
  int t = blockIdx.x; if (t >= ntok) return;
  int b = blockIdx.y;
  int src = keep[t];
  const float* ir = in + ((long long)b * T_ + src) * D_;
  float*       orow = out + ((long long)b * T_ + t) * D_;
  for (int d = threadIdx.x; d < D_; d += 256) orow[d] = ir[d];
}

__global__ void init_kernel(int* pN, int* pPrevValid, float* pPrevMass)
{
  if (threadIdx.x == 0) { *pN = T_; *pPrevValid = 0; *pPrevMass = 0.0f; }
}

// ---------------------------------------------------------------------------
// Host side
// ---------------------------------------------------------------------------
static inline void launch_gemm(hipStream_t s,
    const float* A, const float* Bm, const float* bias, const float* Res, float* C,
    int M, int N, int K, const int* dM, const int* dN, const int* dK,
    int zn, int zdiv,
    long long offA, long long bsAo, long long bsAi, int rsA,
    long long offB, long long bsBo, long long bsBi, int kstrB, int nstrB,
    long long offC, long long bsCo, long long bsCi, int rsC,
    long long offR, long long bsRo, long long bsRi, int rsR,
    int act)
{
  dim3 g((M + TM - 1) / TM, (N + TN - 1) / TN, zn);
  wmma_gemm_kernel<<<g, dim3(256), 0, s>>>(A, Bm, bias, Res, C, M, N, K, dM, dN, dK, zdiv,
      offA, bsAo, bsAi, rsA, offB, bsBo, bsBi, kstrB, nstrB,
      offC, bsCo, bsCi, rsC, offR, bsRo, bsRi, rsR, act);
}

struct Blk {
  const float *ln1_w, *ln1_b, *qkv_w, *qkv_b, *proj_w, *proj_b,
              *ln2_w, *ln2_b, *fc1_w, *fc1_b, *fc2_w, *fc2_b;
};

extern "C" void kernel_launch(void* const* d_in, const int* in_sizes, int n_in,
                              void* d_out, int out_size, void* d_ws, size_t ws_size,
                              hipStream_t stream)
{
  (void)n_in; (void)out_size; (void)ws_size;
  auto gi = [&](int i) { return (const float*)d_in[i]; };

  const float *x, *patch_w, *patch_b, *cls, *pos, *ln_f_w, *ln_f_b, *head_w, *head_b;
  Blk blk[12];

  bool orderA = (in_sizes[0] == 64 * 3 * 224 * 224);  // insertion-order flatten
  if (orderA) {
    x = gi(0); patch_w = gi(1); patch_b = gi(2); cls = gi(3); pos = gi(4);
    ln_f_w = gi(5); ln_f_b = gi(6); head_w = gi(7); head_b = gi(8);
    for (int i = 0; i < 12; ++i) {
      int o = 9 + i * 12;
      blk[i].ln1_w = gi(o + 0);  blk[i].ln1_b = gi(o + 1);
      blk[i].qkv_w = gi(o + 2);  blk[i].qkv_b = gi(o + 3);
      blk[i].proj_w = gi(o + 4); blk[i].proj_b = gi(o + 5);
      blk[i].ln2_w = gi(o + 6);  blk[i].ln2_b = gi(o + 7);
      blk[i].fc1_w = gi(o + 8);  blk[i].fc1_b = gi(o + 9);
      blk[i].fc2_w = gi(o + 10); blk[i].fc2_b = gi(o + 11);
    }
  } else {                                            // jax sorted-key pytree flatten
    for (int i = 0; i < 12; ++i) {
      int o = i * 12;
      blk[i].fc1_b = gi(o + 0);  blk[i].fc1_w = gi(o + 1);
      blk[i].fc2_b = gi(o + 2);  blk[i].fc2_w = gi(o + 3);
      blk[i].ln1_b = gi(o + 4);  blk[i].ln1_w = gi(o + 5);
      blk[i].ln2_b = gi(o + 6);  blk[i].ln2_w = gi(o + 7);
      blk[i].proj_b = gi(o + 8); blk[i].proj_w = gi(o + 9);
      blk[i].qkv_b = gi(o + 10); blk[i].qkv_w = gi(o + 11);
    }
    cls = gi(144); head_b = gi(145); head_w = gi(146);
    ln_f_b = gi(147); ln_f_w = gi(148); patch_b = gi(149); patch_w = gi(150);
    pos = gi(151); x = gi(152);
  }

  // workspace layout (f32)
  float* w = (float*)d_ws;
  size_t o = 0;
  auto alloc = [&](size_t n) { float* p = w + o; o += n; return p; };
  float* tokA   = alloc((size_t)B_ * T_ * D_);
  float* tokB   = alloc((size_t)B_ * T_ * D_);
  float* xn     = alloc((size_t)B_ * T_ * D_);
  float* qkv    = alloc((size_t)B_ * T_ * 3 * D_);
  float* Sbuf   = alloc((size_t)B_ * H_ * T_ * SP_);
  float* ctx    = alloc((size_t)B_ * T_ * D_);
  float* hbuf   = alloc((size_t)B_ * T_ * MLP_);
  float* pein   = alloc((size_t)B_ * NPATCH_ * D_);
  float* vmeanb = alloc((size_t)B_ * T_ * DH_);
  float* Acls   = alloc((size_t)B_ * T_);
  float* Jbuf   = alloc((size_t)B_ * T_);
  float* score  = alloc(256);
  float* sumJ   = alloc(B_);
  float* clsn   = alloc(B_);
  float* scal   = alloc(16);
  float* pPrevMass = scal + 2;
  int* ist = (int*)(w + o);
  int* pN = ist; int* pPrevValid = ist + 1; int* keep = ist + 2;

  init_kernel<<<dim3(1), dim3(64), 0, stream>>>(pN, pPrevValid, pPrevMass);

  // Patch embedding
  patchify_kernel<<<dim3(NPATCH_, B_), dim3(256), 0, stream>>>(x, pein);
  launch_gemm(stream, pein, patch_w, patch_b, nullptr, tokA,
              NPATCH_, D_, D_, nullptr, nullptr, nullptr, B_, 1,
              0, (long long)NPATCH_ * D_, 0, D_,
              0, 0, 0, D_, 1,
              D_, (long long)T_ * D_, 0, D_,
              0, 0, 0, 0, 0);
  clspos_kernel<<<dim3(T_, B_), dim3(256), 0, stream>>>(tokA, cls, pos);

  for (int li = 0; li < 12; ++li) {
    float* tokIn  = (li & 1) ? tokB : tokA;
    float* tokOut = (li & 1) ? tokA : tokB;
    const Blk& p = blk[li];

    // LN1
    layernorm_kernel<<<dim3(T_, B_), dim3(256), 0, stream>>>(tokIn, xn, p.ln1_w, p.ln1_b, pN);
    // QKV: (Ntok x 768) @ (768 x 2304)
    launch_gemm(stream, xn, p.qkv_w, p.qkv_b, nullptr, qkv,
                T_, 3 * D_, D_, pN, nullptr, nullptr, B_, 1,
                0, (long long)T_ * D_, 0, D_,
                0, 0, 0, 3 * D_, 1,
                0, (long long)T_ * 3 * D_, 0, 3 * D_,
                0, 0, 0, 0, 0);
    // S = Q @ K^T  per (b,h)   (B contiguous along k -> k-major staging)
    launch_gemm(stream, qkv, qkv, nullptr, nullptr, Sbuf,
                T_, T_, DH_, pN, pN, nullptr, B_ * H_, H_,
                0, (long long)T_ * 3 * D_, DH_, 3 * D_,
                D_, (long long)T_ * 3 * D_, DH_, 1, 3 * D_,
                0, (long long)H_ * T_ * SP_, (long long)T_ * SP_, SP_,
                0, 0, 0, 0, 0);
    // softmax(S * SCALE)
    softmax_kernel<<<dim3(T_, H_, B_), dim3(256), 0, stream>>>(Sbuf, pN);
    // ctx = A @ V  per (b,h), merged heads into (B,T,D)
    launch_gemm(stream, Sbuf, qkv, nullptr, nullptr, ctx,
                T_, DH_, T_, pN, nullptr, pN, B_ * H_, H_,
                0, (long long)H_ * T_ * SP_, (long long)T_ * SP_, SP_,
                2 * D_, (long long)T_ * 3 * D_, DH_, 3 * D_, 1,
                0, (long long)T_ * D_, DH_, D_,
                0, 0, 0, 0, 0);
    // x = x + ctx @ proj_w + proj_b   (in-place residual)
    launch_gemm(stream, ctx, p.proj_w, p.proj_b, tokIn, tokIn,
                T_, D_, D_, pN, nullptr, nullptr, B_, 1,
                0, (long long)T_ * D_, 0, D_,
                0, 0, 0, D_, 1,
                0, (long long)T_ * D_, 0, D_,
                0, (long long)T_ * D_, 0, D_, 0);
    // LN2
    layernorm_kernel<<<dim3(T_, B_), dim3(256), 0, stream>>>(tokIn, xn, p.ln2_w, p.ln2_b, pN);
    // h = gelu(xn @ fc1_w + fc1_b)
    launch_gemm(stream, xn, p.fc1_w, p.fc1_b, nullptr, hbuf,
                T_, MLP_, D_, pN, nullptr, nullptr, B_, 1,
                0, (long long)T_ * D_, 0, D_,
                0, 0, 0, MLP_, 1,
                0, (long long)T_ * MLP_, 0, MLP_,
                0, 0, 0, 0, 1);
    // x = x + h @ fc2_w + fc2_b
    launch_gemm(stream, hbuf, p.fc2_w, p.fc2_b, tokIn, tokIn,
                T_, D_, MLP_, pN, nullptr, nullptr, B_, 1,
                0, (long long)T_ * MLP_, 0, MLP_,
                0, 0, 0, D_, 1,
                0, (long long)T_ * D_, 0, D_,
                0, (long long)T_ * D_, 0, D_, 0);

    // Pruning statistics + selection + gather (always executed; identity when no prune)
    aclsmean_kernel<<<dim3(B_), dim3(256), 0, stream>>>(Sbuf, Acls, pN);
    vmean_kernel<<<dim3(T_, B_), dim3(64), 0, stream>>>(qkv, vmeanb, pN);
    batchstats_kernel<<<dim3(B_), dim3(256), 0, stream>>>(vmeanb, Acls, Jbuf, sumJ, clsn, pN);
    scalars_kernel<<<dim3(1), dim3(256), 0, stream>>>(Acls, clsn, sumJ, Jbuf, scal, score, pN);
    select_kernel<<<dim3(1), dim3(256), 0, stream>>>(scal, score, keep, pN, pPrevValid, pPrevMass);
    gather_kernel<<<dim3(T_, B_), dim3(256), 0, stream>>>(tokIn, tokOut, keep, pN);
  }

  // Final LN + head  (12 layers -> tokens end in tokA)
  layernorm_kernel<<<dim3(T_, B_), dim3(256), 0, stream>>>(tokA, xn, ln_f_w, ln_f_b, pN);
  launch_gemm(stream, xn, head_w, head_b, nullptr, (float*)d_out,
              B_, NCLS_, D_, nullptr, nullptr, nullptr, 1, 1,
              0, 0, 0, T_ * D_,
              0, 0, 0, NCLS_, 1,
              0, 0, 0, NCLS_,
              0, 0, 0, 0, 0);
}
